// SleepTransformer_75565654606287
// MI455X (gfx1250) — matrix operand order repaired and checked by reading, not compiled, not run on hardware
//
#include <hip/hip_runtime.h>
#include <hip/hip_bf16.h>
#include <math.h>

typedef __bf16 bf16_t;
typedef __attribute__((ext_vector_type(16))) __bf16 v16bf;
typedef __attribute__((ext_vector_type(8)))  __bf16 v8bf;
typedef __attribute__((ext_vector_type(8)))  float  v8f;

// ---------------------------------------------------------------------------
// Weight convert: fp32 W (K x N) -> bf16 Wt (N x K)  (N-major for B fragments)
// ---------------------------------------------------------------------------
__global__ void wconvert_kernel(const float* __restrict__ w, bf16_t* __restrict__ wt,
                                int K, int N, size_t total) {
    size_t i = (size_t)blockIdx.x * blockDim.x + threadIdx.x;
    if (i >= total) return;
    int n = (int)(i / (size_t)K);
    int k = (int)(i % (size_t)K);
    wt[i] = (bf16_t)w[(size_t)k * N + n];
}

// ---------------------------------------------------------------------------
// Embedding: y = x * sqrt(128) + pe[row % T], write fp32 master + bf16 copy
// ---------------------------------------------------------------------------
__global__ void embed_kernel(const float* __restrict__ src, const float* __restrict__ pe,
                             float* __restrict__ x32, bf16_t* __restrict__ x16,
                             int T, size_t total) {
    size_t i = (size_t)blockIdx.x * blockDim.x + threadIdx.x;
    if (i >= total) return;
    int    col = (int)(i & 127);
    size_t row = i >> 7;
    int    t   = (int)(row % (size_t)T);
    float  y   = src[i] * 11.313708498984761f + pe[t * 128 + col];
    x32[i] = y;
    x16[i] = (bf16_t)y;
}

// ---------------------------------------------------------------------------
// BF16 WMMA GEMM with async global->LDS staging of the weight tile.
// C(MxN) = A(MxK) * Wt(NxK)^T + bias, epilogue act, bf16/f32 out.
// block = 256 threads = 8 waves; wave -> 16x64 tile; grid = (M/128, N/64).
// B tile (64 cols x 64 K, 8KB) staged per chunk via GLOBAL_LOAD_ASYNC_TO_LDS,
// double buffered; 72-element padded stride -> conflict-free ds_load_b128.
// B fragments of a k-step preloaded into independent registers so the 4 WMMAs
// are not serialized behind a shared destination octet.
// ---------------------------------------------------------------------------
#define LDS_STRIDE 72                       // elements per staged column (64 + 8 pad)
#define LDS_BUF_ELEMS (64 * LDS_STRIDE)     // 4608 elems = 9216 bytes per buffer

template<int ACT, bool W16, bool W32>
__global__ __launch_bounds__(256)
void gemm_bf16(const bf16_t* __restrict__ A, const bf16_t* __restrict__ Wt,
               const float* __restrict__ bias, bf16_t* __restrict__ C16,
               float* __restrict__ C32, int M, int N, int K) {
    __shared__ __align__(16) bf16_t ldsB[2][LDS_BUF_ELEMS];

    const int tid  = threadIdx.x;
    const int wave = tid >> 5;
    const int lane = tid & 31;
    const int half = lane >> 4;   // K-half selector
    const int idx  = lane & 15;   // row (A) / col (B,C)

    const int row0 = blockIdx.x * 128 + wave * 16;
    const int col0 = blockIdx.y * 64;

    // Staging assignment: thread copies two 16-byte segments per 64-K chunk.
    // Segment g (0..511): column c = g>>3, 16B piece s = g&7. g0 = tid, g1 = tid+256.
    const int      cA  = tid >> 3;           // columns 0..31
    const int      cB  = cA + 32;            // columns 32..63
    const int      sS  = tid & 7;
    const uint32_t ldsBase = (uint32_t)(uintptr_t)&ldsB[0][0];
    const uint32_t lOffA   = (uint32_t)(cA * LDS_STRIDE + sS * 8) * 2u;
    const uint32_t lOffB   = (uint32_t)(cB * LDS_STRIDE + sS * 8) * 2u;
    const uint32_t gOffA   = (uint32_t)(((col0 + cA) * K + sS * 8) * 2);
    const uint32_t gOffB   = (uint32_t)(((col0 + cB) * K + sS * 8) * 2);

    v8f acc[4];
    #pragma unroll
    for (int j = 0; j < 4; ++j)
        #pragma unroll
        for (int e = 0; e < 8; ++e) acc[j][e] = 0.0f;

    const size_t aRow = (size_t)(row0 + idx) * (size_t)K;

    const int nChunks = K >> 6;  // 64-wide K chunks (K is 128 or 1024)

    // prologue: stage chunk 0 into buffer 0
    {
        uint32_t l0 = ldsBase + lOffA, l1 = ldsBase + lOffB;
        uint32_t g0 = gOffA, g1 = gOffB;
        asm volatile("global_load_async_to_lds_b128 %0, %1, %2"
                     :: "v"(l0), "v"(g0), "s"(Wt) : "memory");
        asm volatile("global_load_async_to_lds_b128 %0, %1, %2"
                     :: "v"(l1), "v"(g1), "s"(Wt) : "memory");
    }

    for (int cix = 0; cix < nChunks; ++cix) {
        const int buf = cix & 1;
        const int k0  = cix << 6;

        if (cix + 1 < nChunks) {
            // stage next chunk into the other buffer (its previous readers
            // finished at the barrier that ended the last iteration)
            const uint32_t bufOff = (uint32_t)((buf ^ 1) * (LDS_BUF_ELEMS * 2));
            uint32_t l0 = ldsBase + bufOff + lOffA;
            uint32_t l1 = ldsBase + bufOff + lOffB;
            uint32_t g0 = gOffA + (uint32_t)((k0 + 64) * 2);
            uint32_t g1 = gOffB + (uint32_t)((k0 + 64) * 2);
            asm volatile("global_load_async_to_lds_b128 %0, %1, %2"
                         :: "v"(l0), "v"(g0), "s"(Wt) : "memory");
            asm volatile("global_load_async_to_lds_b128 %0, %1, %2"
                         :: "v"(l1), "v"(g1), "s"(Wt) : "memory");
            asm volatile("s_wait_asynccnt 0x2" ::: "memory");  // chunk cix landed
        } else {
            asm volatile("s_wait_asynccnt 0x0" ::: "memory");
        }
        __syncthreads();   // all waves' staged data visible

        #pragma unroll
        for (int ks = 0; ks < 2; ++ks) {
            const int k = k0 + ks * 32;
            // A fragment: lane holds row (row0+idx); elems 0-7 -> K = k+half*8+e,
            // elems 8-15 -> K = k+16+half*8+e  (ISA 16-bit A layout)
            v16bf a;
            {
                const bf16_t* ap = A + aRow + k + half * 8;
                v8bf lo = *(const v8bf*)ap;
                v8bf hi = *(const v8bf*)(ap + 16);
                #pragma unroll
                for (int e = 0; e < 8; ++e) { a[e] = lo[e]; a[e + 8] = hi[e]; }
            }
            // Preload all 4 B fragments into independent registers so the
            // ds_loads can be clause-issued and the WMMAs overlap the waits.
            v16bf bfrag[4];
            #pragma unroll
            for (int j = 0; j < 4; ++j) {
                // B fragment from LDS: col c = j*16+idx, elems 0-7 -> K = ks*32+half*8+e,
                // elems 8-15 -> K = ks*32+16+half*8+e (same order as A)
                const bf16_t* bp = &ldsB[buf][(j * 16 + idx) * LDS_STRIDE
                                             + ks * 32 + half * 8];
                v8bf lo = *(const v8bf*)bp;
                v8bf hi = *(const v8bf*)(bp + 16);
                #pragma unroll
                for (int e = 0; e < 8; ++e) { bfrag[j][e] = lo[e]; bfrag[j][e + 8] = hi[e]; }
            }
            #pragma unroll
            for (int j = 0; j < 4; ++j) {
                acc[j] = __builtin_amdgcn_wmma_f32_16x16x32_bf16(
                    false, a, false, bfrag[j], (short)0, acc[j], false, false);
            }
        }
        __syncthreads();   // readers done before this buffer is restaged
    }

    #pragma unroll
    for (int j = 0; j < 4; ++j) {
        const int   col = col0 + j * 16 + idx;
        const float bv  = bias[col];
        #pragma unroll
        for (int r = 0; r < 8; ++r) {
            const int row = row0 + half * 8 + r;   // C layout: lane>=16 -> M+8
            float y = acc[j][r] + bv;
            if (ACT == 1) y = fmaxf(y, 0.0f);
            if (ACT == 2) y = tanhf(y);
            const size_t o = (size_t)row * (size_t)N + col;
            if (W16) C16[o] = (bf16_t)y;
            if (W32) C32[o] = y;
        }
    }
}

// ---------------------------------------------------------------------------
// Multi-head attention (T=64, dk=16, d=128): flash-style online softmax.
// grid = (numSeq, 8 heads), block = 64 (one thread per query)
// ---------------------------------------------------------------------------
__global__ __launch_bounds__(64)
void attention_kernel(const bf16_t* __restrict__ q, const bf16_t* __restrict__ k,
                      const bf16_t* __restrict__ v, bf16_t* __restrict__ o) {
    const int    seq  = blockIdx.x;
    const int    h    = blockIdx.y;
    const int    t    = threadIdx.x;
    const size_t base = (size_t)seq * 64 * 128 + (size_t)h * 16;

    float qv[16];
    #pragma unroll
    for (int i = 0; i < 16; ++i) qv[i] = (float)q[base + (size_t)t * 128 + i];

    float m = -1e30f, l = 0.0f, acc[16];
    #pragma unroll
    for (int i = 0; i < 16; ++i) acc[i] = 0.0f;

    for (int u = 0; u < 64; ++u) {
        const bf16_t* kr = k + base + (size_t)u * 128;
        float s = 0.0f;
        #pragma unroll
        for (int i = 0; i < 16; ++i) s += qv[i] * (float)kr[i];
        s *= 0.25f;                       // 1/sqrt(dk=16)
        float mn   = fmaxf(m, s);
        float corr = __expf(m - mn);
        float p    = __expf(s - mn);
        l = l * corr + p;
        const bf16_t* vr = v + base + (size_t)u * 128;
        #pragma unroll
        for (int i = 0; i < 16; ++i) acc[i] = acc[i] * corr + p * (float)vr[i];
        m = mn;
    }
    const float inv = 1.0f / l;
    #pragma unroll
    for (int i = 0; i < 16; ++i)
        o[base + (size_t)t * 128 + i] = (bf16_t)(acc[i] * inv);
}

// ---------------------------------------------------------------------------
// Residual add + LayerNorm over d=128: one wave32 per row (4 elems/lane)
// ---------------------------------------------------------------------------
__global__ __launch_bounds__(256)
void add_ln_kernel(const float* __restrict__ g, const float* __restrict__ res,
                   const float* __restrict__ a, const float* __restrict__ b,
                   float* __restrict__ x32, bf16_t* __restrict__ x16, int M) {
    const int lane = threadIdx.x & 31;
    const int row  = blockIdx.x * 8 + (threadIdx.x >> 5);
    if (row >= M) return;
    const size_t off = (size_t)row * 128;

    float vals[4];
    float s = 0.0f;
    #pragma unroll
    for (int j = 0; j < 4; ++j) {
        int c = lane + j * 32;
        vals[j] = g[off + c] + res[off + c];
        s += vals[j];
    }
    #pragma unroll
    for (int sh = 16; sh > 0; sh >>= 1) s += __shfl_xor(s, sh, 32);
    const float mean = s * (1.0f / 128.0f);

    float vs = 0.0f;
    #pragma unroll
    for (int j = 0; j < 4; ++j) { float d0 = vals[j] - mean; vs += d0 * d0; }
    #pragma unroll
    for (int sh = 16; sh > 0; sh >>= 1) vs += __shfl_xor(vs, sh, 32);
    const float inv = rsqrtf(vs * (1.0f / 128.0f) + 1e-8f);

    #pragma unroll
    for (int j = 0; j < 4; ++j) {
        int   c = lane + j * 32;
        float y = a[c] * (vals[j] - mean) * inv + b[c];
        x32[off + c] = y;
        x16[off + c] = (bf16_t)y;
    }
}

// ---------------------------------------------------------------------------
// Additive attention pooling: scores[t] = vtanh[t] . u ; softmax over T=64;
// pooled[j] = sum_t alpha[t] * hf[t,j].  grid = numSeq, block = 128
// ---------------------------------------------------------------------------
__global__ __launch_bounds__(128)
void attn_pool_kernel(const float* __restrict__ vt /* M x 64 */,
                      const float* __restrict__ hf /* M x 128 */,
                      const float* __restrict__ u,
                      float* __restrict__ pooled /* S x 128 */) {
    __shared__ float sc[64];
    const int seq = blockIdx.x;
    const int tid = threadIdx.x;
    if (tid < 64) {
        const float* vr = vt + ((size_t)seq * 64 + tid) * 64;
        float s = 0.0f;
        for (int aIx = 0; aIx < 64; ++aIx) s += vr[aIx] * u[aIx];
        sc[tid] = s;
    }
    __syncthreads();
    float mx = -1e30f;
    for (int i = 0; i < 64; ++i) mx = fmaxf(mx, sc[i]);
    float sum = 0.0f;
    for (int i = 0; i < 64; ++i) sum += __expf(sc[i] - mx);
    const float inv = 1.0f / sum;
    float acc = 0.0f;
    for (int t = 0; t < 64; ++t)
        acc += __expf(sc[t] - mx) * inv * hf[((size_t)seq * 64 + t) * 128 + tid];
    pooled[(size_t)seq * 128 + tid] = acc;
}

// ---------------------------------------------------------------------------
// Final head: scores = f(Mx1024) @ Wo(1024x5) + bo ; then argmax
// ---------------------------------------------------------------------------
__global__ void fc_out_kernel(const bf16_t* __restrict__ f, const float* __restrict__ Wo,
                              const float* __restrict__ bo, float* __restrict__ scores,
                              int M) {
    int i = blockIdx.x * blockDim.x + threadIdx.x;
    if (i >= M * 5) return;
    int row = i / 5, c = i % 5;
    float s = bo[c];
    const bf16_t* fr = f + (size_t)row * 1024;
    for (int kk = 0; kk < 1024; ++kk) s += (float)fr[kk] * Wo[kk * 5 + c];
    scores[i] = s;
}

__global__ void argmax_kernel(const float* __restrict__ scores,
                              float* __restrict__ pred, int M) {
    int i = blockIdx.x * blockDim.x + threadIdx.x;
    if (i >= M) return;
    const float* s = scores + (size_t)i * 5;
    int best = 0; float bv = s[0];
    for (int c = 1; c < 5; ++c) if (s[c] > bv) { bv = s[c]; best = c; }
    pred[i] = (float)best;
}

// ---------------------------------------------------------------------------
// Host-side orchestration
// ---------------------------------------------------------------------------
template<int ACT, bool W16, bool W32>
static void gemm(const bf16_t* A, const bf16_t* Wt, const float* bias,
                 bf16_t* C16, float* C32, int M, int N, int K, hipStream_t s) {
    dim3 grid(M / 128, N / 64);
    gemm_bf16<ACT, W16, W32><<<grid, 256, 0, s>>>(A, Wt, bias, C16, C32, M, N, K);
}

struct EncBias {
    const float *bq, *bk, *bv, *bo, *ln1a, *ln1b, *b1, *b2, *ln2a, *ln2b;
};

static void run_encoder(int M, const bf16_t* wt_base, const EncBias& p,
                        float* X32, bf16_t* X16, float* G32,
                        bf16_t* Q16, bf16_t* K16, bf16_t* V16, bf16_t* O16,
                        bf16_t* H16, hipStream_t stream) {
    const int d = 128, dff = 1024;
    const size_t WB = 327680;  // bf16 elems per block: 4*128*128 + 2*128*1024
    for (int i = 0; i < 4; ++i) {
        const bf16_t* wq = wt_base + (size_t)i * WB;
        const bf16_t* wk = wq + 16384;
        const bf16_t* wv = wk + 16384;
        const bf16_t* wo = wv + 16384;
        const bf16_t* w1 = wo + 16384;
        const bf16_t* w2 = w1 + 131072;

        gemm<0, true, false>(X16, wq, p.bq + i * d, Q16, nullptr, M, d, d, stream);
        gemm<0, true, false>(X16, wk, p.bk + i * d, K16, nullptr, M, d, d, stream);
        gemm<0, true, false>(X16, wv, p.bv + i * d, V16, nullptr, M, d, d, stream);
        attention_kernel<<<dim3(M / 64, 8), 64, 0, stream>>>(Q16, K16, V16, O16);
        gemm<0, false, true>(O16, wo, p.bo + i * d, nullptr, G32, M, d, d, stream);
        add_ln_kernel<<<M / 8, 256, 0, stream>>>(G32, X32, p.ln1a + i * d,
                                                 p.ln1b + i * d, X32, X16, M);
        gemm<1, true, false>(X16, w1, p.b1 + i * dff, H16, nullptr, M, dff, d, stream);
        gemm<0, false, true>(H16, w2, p.b2 + i * d, nullptr, G32, M, d, dff, stream);
        add_ln_kernel<<<M / 8, 256, 0, stream>>>(G32, X32, p.ln2a + i * d,
                                                 p.ln2b + i * d, X32, X16, M);
    }
}

extern "C" void kernel_launch(void* const* d_in, const int* in_sizes, int n_in,
                              void* d_out, int out_size, void* d_ws, size_t ws_size,
                              hipStream_t stream) {
    (void)in_sizes; (void)n_in; (void)out_size; (void)ws_size;
    // Input order: x, frm_p{Wq,bq,Wk,bk,Wv,bv,Wo,bo,ln1a,ln1b,W1,b1,W2,b2,ln2a,ln2b},
    // attn_p{W,b,u}, seq_p{...}, fc_p{W1,b1,W2,b2,Wo,bo}, frm_pe, seq_pe
    const float* x      = (const float*)d_in[0];
    const float* attnW  = (const float*)d_in[17];
    const float* attnb  = (const float*)d_in[18];
    const float* attnu  = (const float*)d_in[19];
    const float* fcW1   = (const float*)d_in[36];
    const float* fcb1   = (const float*)d_in[37];
    const float* fcW2   = (const float*)d_in[38];
    const float* fcb2   = (const float*)d_in[39];
    const float* fcWo   = (const float*)d_in[40];
    const float* fcbo   = (const float*)d_in[41];
    const float* frm_pe = (const float*)d_in[42];
    const float* seq_pe = (const float*)d_in[43];

    // ---------------- workspace layout ----------------
    char*   ws   = (char*)d_ws;
    bf16_t* WT   = (bf16_t*)(ws);                               // 8 MiB bf16 weights
    float*  X32  = (float*) (ws + ((size_t)8   << 20));         // 64 MiB
    bf16_t* X16  = (bf16_t*)(ws + ((size_t)72  << 20));         // 32 MiB
    float*  G32  = (float*) (ws + ((size_t)104 << 20));         // 64 MiB
    char*   AREA =          ws + ((size_t)168 << 20);           // 256 MiB arena
    float*  P32  = (float*) (ws + ((size_t)424 << 20));         // 1 MiB pooled

    bf16_t* Q16 = (bf16_t*)(AREA);
    bf16_t* K16 = (bf16_t*)(AREA + ((size_t)32 << 20));
    bf16_t* V16 = (bf16_t*)(AREA + ((size_t)64 << 20));
    bf16_t* O16 = (bf16_t*)(AREA + ((size_t)96 << 20));
    bf16_t* H16 = (bf16_t*)(AREA);                   // reuses QKV/O lifetime
    bf16_t* F1  = (bf16_t*)(AREA);                   // FC buffers after encoders
    bf16_t* F2  = (bf16_t*)(AREA + ((size_t)8 << 20));

    const size_t ENC_WT  = 1310720;          // bf16 elems per encoder
    bf16_t* WT_frm  = WT;
    bf16_t* WT_seq  = WT + ENC_WT;
    bf16_t* WT_attn = WT + 2 * ENC_WT;       // 64 x 128
    bf16_t* WT_fc1  = WT_attn + 8192;        // 1024 x 128
    bf16_t* WT_fc2  = WT_fc1 + 131072;       // 1024 x 1024

    auto conv = [&](const float* src, bf16_t* dst, int K, int N) {
        size_t tot = (size_t)K * N;
        wconvert_kernel<<<(unsigned)((tot + 255) / 256), 256, 0, stream>>>(src, dst, K, N, tot);
    };

    // ---------------- weight conversion ----------------
    const size_t WB = 327680;
    for (int e = 0; e < 2; ++e) {
        const int base = (e == 0) ? 1 : 20;
        bf16_t* wt = (e == 0) ? WT_frm : WT_seq;
        const float* Wq = (const float*)d_in[base + 0];
        const float* Wk = (const float*)d_in[base + 2];
        const float* Wv = (const float*)d_in[base + 4];
        const float* Wo = (const float*)d_in[base + 6];
        const float* W1 = (const float*)d_in[base + 10];
        const float* W2 = (const float*)d_in[base + 12];
        for (int i = 0; i < 4; ++i) {
            conv(Wq + (size_t)i * 16384,  wt + (size_t)i * WB + 0,      128, 128);
            conv(Wk + (size_t)i * 16384,  wt + (size_t)i * WB + 16384,  128, 128);
            conv(Wv + (size_t)i * 16384,  wt + (size_t)i * WB + 32768,  128, 128);
            conv(Wo + (size_t)i * 16384,  wt + (size_t)i * WB + 49152,  128, 128);
            conv(W1 + (size_t)i * 131072, wt + (size_t)i * WB + 65536,  128, 1024);
            conv(W2 + (size_t)i * 131072, wt + (size_t)i * WB + 196608, 1024, 128);
        }
    }
    conv(attnW, WT_attn, 128, 64);
    conv(fcW1,  WT_fc1,  128, 1024);
    conv(fcW2,  WT_fc2,  1024, 1024);

    auto bias_set = [&](int base) {
        EncBias p;
        p.bq   = (const float*)d_in[base + 1];
        p.bk   = (const float*)d_in[base + 3];
        p.bv   = (const float*)d_in[base + 5];
        p.bo   = (const float*)d_in[base + 7];
        p.ln1a = (const float*)d_in[base + 8];
        p.ln1b = (const float*)d_in[base + 9];
        p.b1   = (const float*)d_in[base + 11];
        p.b2   = (const float*)d_in[base + 13];
        p.ln2a = (const float*)d_in[base + 14];
        p.ln2b = (const float*)d_in[base + 15];
        return p;
    };

    // ---------------- frame-level transformer (M = 32*64*64 = 131072) ----------------
    const int Mf = 131072;
    {
        size_t tot = (size_t)Mf * 128;
        embed_kernel<<<(unsigned)((tot + 255) / 256), 256, 0, stream>>>(
            x, frm_pe, X32, X16, 64, tot);
    }
    EncBias pf = bias_set(1);
    run_encoder(Mf, WT_frm, pf, X32, X16, G32, Q16, K16, V16, O16, H16, stream);

    // ---------------- attention pooling over Tf ----------------
    gemm<2, false, true>(X16, WT_attn, attnb, nullptr, G32, Mf, 64, 128, stream);
    attn_pool_kernel<<<2048, 128, 0, stream>>>(G32, X32, attnu, P32);

    // ---------------- sequence-level transformer (M = 32*64 = 2048) ----------------
    const int Ms = 2048;
    {
        size_t tot = (size_t)Ms * 128;
        embed_kernel<<<(unsigned)((tot + 255) / 256), 256, 0, stream>>>(
            P32, seq_pe, X32, X16, 64, tot);
    }
    EncBias ps = bias_set(20);
    run_encoder(Ms, WT_seq, ps, X32, X16, G32, Q16, K16, V16, O16, H16, stream);

    // ---------------- FC head ----------------
    gemm<1, true, false>(X16, WT_fc1, fcb1, F1, nullptr, Ms, 1024, 128, stream);
    gemm<1, true, false>(F1,  WT_fc2, fcb2, F2, nullptr, Ms, 1024, 1024, stream);

    float* scores = (float*)d_out;                 // (32, 64, 5) flat
    float* pred   = scores + (size_t)Ms * 5;       // (32, 64) as float
    fc_out_kernel<<<(Ms * 5 + 255) / 256, 256, 0, stream>>>(F2, fcWo, fcbo, scores, Ms);
    argmax_kernel<<<(Ms + 255) / 256, 256, 0, stream>>>(scores, pred, Ms);
}
